// Encoder_18365280157999
// MI455X (gfx1250) — matrix operand order repaired and checked by reading, not compiled, not run on hardware
//
#include <hip/hip_runtime.h>

typedef __attribute__((ext_vector_type(16))) __bf16 v16bf;
typedef __attribute__((ext_vector_type(8)))  __bf16 v8bf;
typedef __attribute__((ext_vector_type(8)))  float  v8f;

#define KDIM 256
#define KTILES 8

__device__ __forceinline__ unsigned short f2bf(float f) {
  union { float f; unsigned int u; } v; v.f = f;
  unsigned int r = v.u + 0x7FFFu + ((v.u >> 16) & 1u);   // round-to-nearest-even
  return (unsigned short)(r >> 16);
}

// ---------------- degree / normalization ----------------
__global__ void k_init_deg(float* __restrict__ deg, int n) {
  int i = blockIdx.x * blockDim.x + threadIdx.x;
  if (i < n) deg[i] = 1.0f;                    // self-loop weight baked in
}

__global__ void k_deg(const int* __restrict__ col, const float* __restrict__ ew,
                      float* __restrict__ deg, int ne) {
  int i = blockIdx.x * blockDim.x + threadIdx.x;
  if (i < ne) unsafeAtomicAdd(&deg[col[i]], ew[i]);
}

__global__ void k_dinv(const float* __restrict__ deg, float* __restrict__ dinv, int n) {
  int i = blockIdx.x * blockDim.x + threadIdx.x;
  if (i < n) dinv[i] = rsqrtf(deg[i]);         // deg >= 1 always
}

// ---------------- fp32 -> bf16 (vectorized) ----------------
__global__ void k_cvt_bf16(const float* __restrict__ in, unsigned short* __restrict__ out,
                           long n4) {
  long i = (long)blockIdx.x * blockDim.x + threadIdx.x;
  if (i >= n4) return;
  float4 v = ((const float4*)in)[i];
  ushort4 o;
  o.x = f2bf(v.x); o.y = f2bf(v.y); o.z = f2bf(v.z); o.w = f2bf(v.w);
  ((ushort4*)out)[i] = o;
}

// ---------------- weight pre-swizzle into WMMA B-fragment layout ----------------
// out[((nt*KTILES + kt)*32 + lane)*16 + e] = bf16(W[k*ndim + n])
//   n = nt*16 + lane%16 ; k = kt*32 + (lane<16 ? 0 : 8) + (e<8 ? e : e+8)
__global__ void k_swizzle(const float* __restrict__ W, unsigned short* __restrict__ out,
                          int ndim) {
  int idx = blockIdx.x * blockDim.x + threadIdx.x;
  int total = KDIM * ndim;
  if (idx >= total) return;
  int e    = idx & 15;
  int lane = (idx >> 4) & 31;
  int kt   = (idx >> 9) & 7;
  int nt   = idx >> 12;
  int n = nt * 16 + (lane & 15);
  int k = kt * 32 + ((lane < 16) ? 0 : 8) + ((e < 8) ? e : e + 8);
  out[idx] = f2bf(W[k * ndim + n]);
}

// ---------------- WMMA GEMM: C[nrows, ndim] = A[nrows, 256] @ B ----------------
// A strip (16x256, 64 VGPRs/lane) is preloaded ONCE per strip and held in
// registers; column tiles processed 2 at a time (16 acc VGPRs) -> no spills.
// MODE 0: bf16 out, +bias, relu   (dense layer)
// MODE 1: f32 out, no bias        (pre-aggregation linear)
// MODE 2: f32 out, +bias          (mu / logstd heads)
template <int MODE>
__global__ void k_gemm(const unsigned short* __restrict__ A,
                       const unsigned short* __restrict__ Bsw,
                       const float* __restrict__ bias,
                       void* __restrict__ Cout, int nrows, int ndim) {
  const int wave  = threadIdx.x >> 5;
  const int lane  = threadIdx.x & 31;
  const int strip = blockIdx.x * (blockDim.x >> 5) + wave;
  if (strip * 16 >= nrows) return;
  const int row0  = strip * 16;
  const int hi    = (lane >= 16) ? 8 : 0;   // A: K sub-half select; C: M += 8
  const int mlane = lane & 15;
  const unsigned short* arow = A + (size_t)(row0 + mlane) * KDIM;
  const int ntiles = ndim >> 4;

  // Preload entire A strip: lane<16 holds K {0..7,16..23}, lane>=16 {8..15,24..31}
  v16bf af[KTILES];
#pragma unroll
  for (int kt = 0; kt < KTILES; ++kt) {
    const unsigned short* ap = arow + kt * 32 + hi;
    v8bf alo = *(const v8bf*)(ap);
    v8bf ahi = *(const v8bf*)(ap + 16);
    af[kt] = __builtin_shufflevector(alo, ahi,
              0,1,2,3,4,5,6,7,8,9,10,11,12,13,14,15);
  }

  for (int nt0 = 0; nt0 < ntiles; nt0 += 2) {
    v8f acc0 = {}, acc1 = {};
#pragma unroll
    for (int kt = 0; kt < KTILES; ++kt) {
      const unsigned short* bbase =
          Bsw + (((size_t)nt0 * KTILES + kt) * 32 + lane) * 16;
      v16bf b0 = *(const v16bf*)(bbase);
      v16bf b1 = *(const v16bf*)(bbase + (KTILES * 32 * 16));
      acc0 = __builtin_amdgcn_wmma_f32_16x16x32_bf16(
          false, af[kt], false, b0, (short)0, acc0, false, false);
      acc1 = __builtin_amdgcn_wmma_f32_16x16x32_bf16(
          false, af[kt], false, b1, (short)0, acc1, false, false);
    }
    // C/D layout: VGPR r -> M = r + (lane<16 ? 0 : 8), N = nt*16 + lane%16
#pragma unroll
    for (int c = 0; c < 2; ++c) {
      int ncol = (nt0 + c) * 16 + mlane;
      float bv = 0.f;
      if (MODE != 1) bv = bias[ncol];
      const v8f& acc = (c == 0) ? acc0 : acc1;
#pragma unroll
      for (int r = 0; r < 8; ++r) {
        float v = acc[r] + bv;
        size_t oidx = (size_t)(row0 + hi + r) * ndim + ncol;
        if (MODE == 0) {
          v = v > 0.f ? v : 0.f;
          ((unsigned short*)Cout)[oidx] = f2bf(v);
        } else {
          ((float*)Cout)[oidx] = v;
        }
      }
    }
  }
}

// ---------------- aggregation: init with self-loop term (+ optional bias) -------
__global__ void k_agg_init(const float* __restrict__ t, const float* __restrict__ dinv,
                           const float* __restrict__ bias, float* __restrict__ out,
                           long total) {
  long i = (long)blockIdx.x * blockDim.x + threadIdx.x;
  if (i >= total) return;
  int node = (int)(i >> 8);
  int j    = (int)(i & 255);
  float d  = dinv[node];
  float v  = d * d * t[i];
  if (bias) v += bias[j];
  out[i] = v;
}

// ---------------- aggregation: edge scatter (1 wave/edge, 8 floats/lane) --------
__global__ void k_agg_edges(const float* __restrict__ h, const int* __restrict__ row,
                            const int* __restrict__ col, const float* __restrict__ ew,
                            const float* __restrict__ dinv, float* __restrict__ out,
                            int ne) {
  int e = blockIdx.x * (blockDim.x >> 5) + (threadIdx.x >> 5);
  if (e >= ne) return;
  int lane = threadIdx.x & 31;
  int r = row[e], c = col[e];
  float coef = dinv[r] * ew[e] * dinv[c];
  const float* hp = h + (size_t)r * 256 + lane * 8;
  float* op = out + (size_t)c * 256 + lane * 8;
  float4 a = *(const float4*)hp;
  float4 b = *(const float4*)(hp + 4);
  unsafeAtomicAdd(op + 0, a.x * coef);
  unsafeAtomicAdd(op + 1, a.y * coef);
  unsafeAtomicAdd(op + 2, a.z * coef);
  unsafeAtomicAdd(op + 3, a.w * coef);
  unsafeAtomicAdd(op + 4, b.x * coef);
  unsafeAtomicAdd(op + 5, b.y * coef);
  unsafeAtomicAdd(op + 6, b.z * coef);
  unsafeAtomicAdd(op + 7, b.w * coef);
}

extern "C" void kernel_launch(void* const* d_in, const int* in_sizes, int n_in,
                              void* d_out, int out_size, void* d_ws, size_t ws_size,
                              hipStream_t stream) {
  const float* x       = (const float*)d_in[0];
  const int*   eidx    = (const int*)d_in[1];
  const float* ew      = (const float*)d_in[2];
  const float* w_dense = (const float*)d_in[3];
  const float* b_dense = (const float*)d_in[4];
  const float* w_enc   = (const float*)d_in[5];
  const float* b_enc   = (const float*)d_in[6];
  const float* w_mu    = (const float*)d_in[7];
  const float* b_mu    = (const float*)d_in[8];
  const float* w_ls    = (const float*)d_in[9];
  const float* b_ls    = (const float*)d_in[10];

  const int N  = in_sizes[0] / 256;
  const int E_ = in_sizes[2];
  const int* row = eidx;
  const int* col = eidx + E_;
  const long NF = (long)N * 256;

  // workspace carve-out (~308 MB with reuse)
  char* ws = (char*)d_ws;
  auto carve = [&](size_t bytes) -> char* {
    char* p = ws; ws += (bytes + 255) & ~(size_t)255; return p;
  };
  float* deg            = (float*)carve((size_t)N * 4);
  float* dinv           = (float*)carve((size_t)N * 4);
  unsigned short* swzd  = (unsigned short*)carve(256 * 256 * 2);
  unsigned short* swze  = (unsigned short*)carve(256 * 256 * 2);
  unsigned short* swzm  = (unsigned short*)carve(256 * 128 * 2);
  unsigned short* swzl  = (unsigned short*)carve(256 * 128 * 2);
  unsigned short* xbf   = (unsigned short*)carve((size_t)NF * 2);  // reused as a2_bf16
  unsigned short* h0bf  = (unsigned short*)carve((size_t)NF * 2);
  float* t1             = (float*)carve((size_t)NF * 4);           // reused as a2
  float* h1             = (float*)carve((size_t)NF * 4);

  // degree + D^-1/2
  k_init_deg<<<(N + 255) / 256, 256, 0, stream>>>(deg, N);
  k_deg<<<(E_ + 255) / 256, 256, 0, stream>>>(col, ew, deg, E_);
  k_dinv<<<(N + 255) / 256, 256, 0, stream>>>(deg, dinv, N);

  // operand conversion / weight swizzle
  k_cvt_bf16<<<(int)((NF / 4 + 255) / 256), 256, 0, stream>>>(x, xbf, NF / 4);
  k_swizzle<<<(256 * 256 + 255) / 256, 256, 0, stream>>>(w_dense, swzd, 256);
  k_swizzle<<<(256 * 256 + 255) / 256, 256, 0, stream>>>(w_enc,   swze, 256);
  k_swizzle<<<(256 * 128 + 255) / 256, 256, 0, stream>>>(w_mu,    swzm, 128);
  k_swizzle<<<(256 * 128 + 255) / 256, 256, 0, stream>>>(w_ls,    swzl, 128);

  const int strips  = (N + 15) / 16;
  const int gblocks = (strips + 7) / 8;   // 8 waves (16-row strips) per block

  // h0 = relu(x @ W_dense + b_dense)  [bf16 out]
  k_gemm<0><<<gblocks, 256, 0, stream>>>(xbf, swzd, b_dense, h0bf, N, 256);
  // t1 = h0 @ W_enc                   [f32 out]
  k_gemm<1><<<gblocks, 256, 0, stream>>>(h0bf, swze, nullptr, t1, N, 256);

  // h1 = Â t1 + b_enc
  k_agg_init<<<(int)((NF + 255) / 256), 256, 0, stream>>>(t1, dinv, b_enc, h1, NF);
  k_agg_edges<<<(E_ + 7) / 8, 256, 0, stream>>>(t1, row, col, ew, dinv, h1, E_);

  // a2 = Â h1   (shared by mu & logstd since Â(hW) = (Âh)W)
  float* a2 = t1;
  k_agg_init<<<(int)((NF + 255) / 256), 256, 0, stream>>>(h1, dinv, nullptr, a2, NF);
  k_agg_edges<<<(E_ + 7) / 8, 256, 0, stream>>>(h1, row, col, ew, dinv, a2, E_);

  unsigned short* a2bf = xbf;
  k_cvt_bf16<<<(int)((NF / 4 + 255) / 256), 256, 0, stream>>>(a2, a2bf, NF / 4);

  // mu / logstd heads -> d_out (mu first, then logstd)
  float* mu = (float*)d_out;
  float* ls = mu + (size_t)N * 128;
  k_gemm<2><<<gblocks, 256, 0, stream>>>(a2bf, swzm, b_mu, mu, N, 128);
  k_gemm<2><<<gblocks, 256, 0, stream>>>(a2bf, swzl, b_ls, ls, N, 128);
}
